// ESNPCA_35373350650550
// MI455X (gfx1250) — matrix operand order repaired and checked by reading, not compiled, not run on hardware
//
#include <hip/hip_runtime.h>
#include <hip/hip_bf16.h>
#include <math.h>

// Problem constants (from reference)
#define R      2048
#define NI     8
#define NT     16384
#define OUTW   (R + NI)     // 2056
#define LRATE  0.1f

// Decomposition
#define NB          64            // persistent workgroups
#define ROWS_PER_WG (R / NB)      // 32 rows of Wh per WG
#define BDIM        64            // 2 waves (wave32) per WG, 1 x 16-row WMMA tile each
#define KC          (R / 32)      // 64 K-chunks of 32 (bf16 WMMA K)

typedef __attribute__((ext_vector_type(16))) __bf16 v16bf;
typedef __attribute__((ext_vector_type(8)))  float  v8f;

union BF16x16 { v16bf v; __bf16 e[16]; };
union F32x8   { v8f   v; float  e[8];  };

// ---------------------------------------------------------------------------
// Init: H tail columns get x_t, h double-buffer slot 0 gets h0, barrier = 0.
// ---------------------------------------------------------------------------
__global__ void esn_init(const float* __restrict__ x, const float* __restrict__ h0,
                         float* __restrict__ H, float* __restrict__ hbuf,
                         unsigned* __restrict__ cnt) {
  int idx = blockIdx.x * blockDim.x + threadIdx.x;
  if (idx < NT * NI) {
    int t = idx >> 3, i = idx & 7;
    H[(size_t)t * OUTW + R + i] = x[idx];
  }
  if (idx < R) hbuf[idx] = h0[idx];
  if (idx == 0) *cnt = 0u;
}

// ---------------------------------------------------------------------------
// Persistent recurrence kernel.
// LDS layout:
//   aFrag : [2 waves][KC][32 lanes] x v16bf  (Wh slice in WMMA A-frag order, bf16)
//   hlds  : R x bf16 (current h, staged each step)
// ---------------------------------------------------------------------------
__launch_bounds__(BDIM, 1)
__global__ void esn_recur(const float* __restrict__ x, const float* __restrict__ Win,
                          const float* __restrict__ Wh, float* __restrict__ H,
                          float* __restrict__ hbuf, unsigned* __restrict__ cnt) {
  extern __shared__ char smem[];
  v16bf*  aFrag = (v16bf*)smem;                            // 2*KC*32*32B = 128KB
  __bf16* hlds  = (__bf16*)(smem + (size_t)2 * KC * 32 * 32);  // +4KB

  const int tid  = threadIdx.x;
  const int wave = tid >> 5;
  const int lane = tid & 31;
  const int wg   = blockIdx.x;
  const int rowBase = wg * ROWS_PER_WG + wave * 16;

  // ---- one-time: swizzle this WG's Wh rows into WMMA A-fragment order ----
  // A order used here (not AMD's table order): lane L holds row (L&15); elems 0..15 = K of
  // kbase + (L>>4)*16 .. +15 (contiguous, matching B)
  {
    const int   lrow  = lane & 15;
    const int   khalf = (lane >> 4) * 16;
    const float* wrow = Wh + (size_t)(rowBase + lrow) * R;
    for (int c = 0; c < KC; ++c) {
      BF16x16 fa;
      const int k0 = c * 32 + khalf;
#pragma unroll
      for (int e = 0; e < 8; ++e) fa.e[e]     = (__bf16)wrow[k0 + e];
#pragma unroll
      for (int e = 0; e < 8; ++e) fa.e[8 + e] = (__bf16)wrow[k0 + 8 + e];
      aFrag[((size_t)wave * KC + c) * 32 + lane] = fa.v;
    }
  }

  // C-frag layout: VGPR v = row v (lanes 0-15) / row v+8 (lanes 16-31).
  // Writer lanes 0..7 own rows 0..7 (col 0), lanes 16..23 own rows 8..15.
  const bool writer = ((lane & 15) < 8);
  const int  lRow   = (lane & 15) + ((lane >> 4) << 3);
  const int  gRow   = rowBase + lRow;
  float wr[NI];
#pragma unroll
  for (int i = 0; i < NI; ++i) wr[i] = Win[gRow * NI + i];

  __syncthreads();

  const v16bf* aw   = aFrag + (size_t)wave * KC * 32;
  const int    boff = (lane >> 4) * 16;   // B 32x16 replicated-h: 16 contig bf16 / lane-half

  for (int t = 0; t < NT; ++t) {
    const float* hin  = hbuf + (size_t)(t & 1) * R;
    float*       hout = hbuf + (size_t)((t + 1) & 1) * R;

    // stage h (fp32 -> bf16) into LDS
    for (int i = tid; i < R; i += BDIM) hlds[i] = (__bf16)hin[i];
    if (tid == 0) __builtin_prefetch(x + (size_t)(t + 1) * NI, 0, 1);
    __syncthreads();

    // 16 rows x K=2048 via 64 chained bf16 WMMAs (fp32 accumulate)
    F32x8 acc; acc.v = (v8f){};
    for (int c = 0; c < KC; ++c) {
      v16bf a = aw[(size_t)c * 32 + lane];
      v16bf b = *(const v16bf*)(hlds + c * 32 + boff);
      acc.v = __builtin_amdgcn_wmma_f32_16x16x32_bf16(
          /*neg_a=*/false, a, /*neg_b=*/false, b,
          /*c_mod=*/(short)0, acc.v, /*reuse_a=*/false, /*reuse_b=*/false);
    }

    if (writer) {
      const float* xt = x + (size_t)t * NI;
      float u = 0.f;
#pragma unroll
      for (int i = 0; i < NI; ++i) u = fmaf(wr[i], xt[i], u);
      float hp = hin[gRow];
      float hn = tanhf(acc.e[lane & 7] + u);
      float ht = (1.0f - LRATE) * hp + LRATE * hn;
      hout[gRow] = ht;
      H[(size_t)t * OUTW + gRow] = ht;
    }
    __syncthreads();

    // grid-wide barrier: monotonic counter, release-add / acquire-spin
    if (tid == 0) {
      __threadfence();
      __hip_atomic_fetch_add(cnt, 1u, __ATOMIC_RELEASE, __HIP_MEMORY_SCOPE_AGENT);
      const unsigned target = (unsigned)(t + 1) * (unsigned)NB;
      while (__hip_atomic_load(cnt, __ATOMIC_ACQUIRE, __HIP_MEMORY_SCOPE_AGENT) < target)
        __builtin_amdgcn_s_sleep(2);
    }
    __syncthreads();
    __threadfence();
  }
}

// ---------------------------------------------------------------------------
// Inputs (setup_inputs order): x[T*8], h[2048], Win[2048*8], Wh[2048*2048]
// Output: H[T * 2056]. Workspace: hbuf double buffer (2*R floats) + counter.
// ---------------------------------------------------------------------------
extern "C" void kernel_launch(void* const* d_in, const int* in_sizes, int n_in,
                              void* d_out, int out_size, void* d_ws, size_t ws_size,
                              hipStream_t stream) {
  const float* x   = (const float*)d_in[0];
  const float* h0  = (const float*)d_in[1];
  const float* Win = (const float*)d_in[2];
  const float* Wh  = (const float*)d_in[3];
  float* H    = (float*)d_out;
  float* hbuf = (float*)d_ws;
  unsigned* cnt = (unsigned*)((char*)d_ws + (size_t)2 * R * sizeof(float));

  int initN = NT * NI;  // covers R and the counter too
  esn_init<<<(initN + 255) / 256, 256, 0, stream>>>(x, h0, H, hbuf, cnt);

  size_t smemBytes = (size_t)2 * KC * 32 * 32 /*aFrag*/ + (size_t)R * 2 /*hlds*/;
  esn_recur<<<NB, BDIM, smemBytes, stream>>>(x, Win, Wh, H, hbuf, cnt);
}
